// SAE_62139586839264
// MI455X (gfx1250) — compile-verified, matrix operand order checked by reading, not run on hardware
//
#include <hip/hip_runtime.h>
#include <hip/hip_bf16.h>

typedef __attribute__((ext_vector_type(16))) __bf16 v16bf;
typedef __attribute__((ext_vector_type(8)))  __bf16 v8bf;
typedef __attribute__((ext_vector_type(8)))  float  v8f;
typedef __attribute__((ext_vector_type(4)))  int    v4i;

#define TM 128
#define TN 128
#define TK 32
#define TKP 40   // padded LDS row stride (80 B) -> conflict-free fragment reads
#define TOPK 64

#if defined(__AMDGCN__) && __has_builtin(__builtin_amdgcn_global_load_async_to_lds_b128)
#define USE_ASYNC_LDS 1
#else
#define USE_ASYNC_LDS 0
#endif

__device__ __forceinline__ void wait_asynccnt0() {
#if defined(__AMDGCN__)
#if __has_builtin(__builtin_amdgcn_s_wait_asynccnt)
  __builtin_amdgcn_s_wait_asynccnt(0);
#else
  asm volatile("s_wait_asynccnt 0x0" ::: "memory");
#endif
#endif
}

#if USE_ASYNC_LDS
__device__ __forceinline__ void async_b128(const void* g, void* l) {
  __builtin_amdgcn_global_load_async_to_lds_b128(
      (__attribute__((address_space(1))) v4i*)g,
      (__attribute__((address_space(3))) v4i*)l, 0, 0);
}
#endif

// ---------------------------------------------------------------------------
// Kernel 1: fp32 -> bf16 elementwise convert (x), 4 elements/thread
// ---------------------------------------------------------------------------
__global__ __launch_bounds__(256) void sae_conv_x(const float* __restrict__ x,
                                                  __bf16* __restrict__ xbf,
                                                  long long n4) {
  long long i = (long long)blockIdx.x * blockDim.x + threadIdx.x;
  if (i >= n4) return;
  float4 v = reinterpret_cast<const float4*>(x)[i];
  __bf16* o = xbf + 4 * i;
  o[0] = (__bf16)v.x;
  o[1] = (__bf16)v.y;
  o[2] = (__bf16)v.z;
  o[3] = (__bf16)v.w;
}

// ---------------------------------------------------------------------------
// Kernel 2: W_enc [D,F] fp32 -> W_encT [F,D] bf16 (transpose + convert)
// ---------------------------------------------------------------------------
__global__ __launch_bounds__(256) void sae_transpose_wenc(const float* __restrict__ W,
                                                          __bf16* __restrict__ Wt,
                                                          int D, int F) {
  __shared__ float t[32][33];
  const int tx = threadIdx.x;       // 0..31
  const int ty0 = threadIdx.y;      // 0..7
  const int f0 = blockIdx.x * 32;
  const int d0 = blockIdx.y * 32;
#pragma unroll
  for (int r = 0; r < 4; ++r) {
    int d = d0 + ty0 + r * 8;
    t[ty0 + r * 8][tx] = W[(size_t)d * F + f0 + tx];
  }
  __syncthreads();
  const int d2 = d0 + tx;
#pragma unroll
  for (int r = 0; r < 4; ++r) {
    int f2 = f0 + ty0 + r * 8;
    Wt[(size_t)f2 * D + d2] = (__bf16)t[tx][ty0 + r * 8];
  }
}

// ---------------------------------------------------------------------------
// Kernel 3: encoder GEMM  z = relu(x @ W_enc + b_enc)
// A = xbf [B,D] bf16, B = W_encT [F,D] bf16 (both K-contiguous).
// 128x128 block tile, K-step 32, 8 waves x (2x4) 16x16 WMMA tiles.
// Tile feed via GLOBAL_LOAD_ASYNC_TO_LDS_B128 (ASYNCcnt) when available.
// ---------------------------------------------------------------------------
__global__ __launch_bounds__(256) void sae_enc_gemm(const __bf16* __restrict__ A,
                                                    const __bf16* __restrict__ Bt,
                                                    const float* __restrict__ bias,
                                                    float* __restrict__ Z,
                                                    int D, int F) {
  __shared__ __align__(16) __bf16 sA[2][TM][TKP];
  __shared__ __align__(16) __bf16 sB[2][TN][TKP];

  const int tid  = threadIdx.x;
  const int lane = tid & 31;
  const int wv   = tid >> 5;
  const int wm   = wv & 3;    // 0..3 : 32-row slab within 128
  const int wn   = wv >> 2;   // 0..1 : 64-col slab within 128
  const int l15  = lane & 15;
  const int khalf = lane >> 4;

  const size_t m0 = (size_t)blockIdx.y * TM;
  const size_t n0 = (size_t)blockIdx.x * TN;
  const int NK = D / TK;

  // tile load mapping: 512 x 16B chunks per tile, 2 chunks/thread
  const int lr = tid >> 2;     // rows 0..63 (and +64)
  const int lq = tid & 3;      // 16B chunk within the 64B of row data

#if USE_ASYNC_LDS
  auto async_tile = [&](int kt, int buf) {
    const __bf16* pa = A + (m0 + lr) * (size_t)D + kt * TK + lq * 8;
    const __bf16* pb = Bt + (n0 + lr) * (size_t)D + kt * TK + lq * 8;
    async_b128(pa,                     &sA[buf][lr][lq * 8]);
    async_b128(pa + (size_t)64 * D,    &sA[buf][lr + 64][lq * 8]);
    async_b128(pb,                     &sB[buf][lr][lq * 8]);
    async_b128(pb + (size_t)64 * D,    &sB[buf][lr + 64][lq * 8]);
  };
#else
  uint4 ra0, ra1, rb0, rb1;
  auto load_regs = [&](int kt) {
    const __bf16* pa = A + (m0 + lr) * (size_t)D + kt * TK + lq * 8;
    ra0 = *reinterpret_cast<const uint4*>(pa);
    ra1 = *reinterpret_cast<const uint4*>(pa + (size_t)64 * D);
    const __bf16* pb = Bt + (n0 + lr) * (size_t)D + kt * TK + lq * 8;
    rb0 = *reinterpret_cast<const uint4*>(pb);
    rb1 = *reinterpret_cast<const uint4*>(pb + (size_t)64 * D);
  };
  auto store_lds = [&](int buf) {
    *reinterpret_cast<uint4*>(&sA[buf][lr][lq * 8])      = ra0;
    *reinterpret_cast<uint4*>(&sA[buf][lr + 64][lq * 8]) = ra1;
    *reinterpret_cast<uint4*>(&sB[buf][lr][lq * 8])      = rb0;
    *reinterpret_cast<uint4*>(&sB[buf][lr + 64][lq * 8]) = rb1;
  };
#endif

  v8f acc[2][4];
#pragma unroll
  for (int sm = 0; sm < 2; ++sm)
#pragma unroll
    for (int sn = 0; sn < 4; ++sn)
#pragma unroll
      for (int i = 0; i < 8; ++i) acc[sm][sn][i] = 0.0f;

#if USE_ASYNC_LDS
  async_tile(0, 0);
  wait_asynccnt0();
  __syncthreads();
#else
  load_regs(0);
  store_lds(0);
  __syncthreads();
#endif

  for (int kt = 0; kt < NK; ++kt) {
    const int buf = kt & 1;
#if USE_ASYNC_LDS
    if (kt + 1 < NK) async_tile(kt + 1, buf ^ 1);
#else
    if (kt + 1 < NK) load_regs(kt + 1);
#endif

    // Fragment loads per the 16-bit A/B VGPR layout:
    // lanes 0-15 hold K 0..7 (lo) and 16..23 (hi); lanes 16-31 hold K 8..15 / 24..31
    v16bf afr[2], bfr[4];
#pragma unroll
    for (int sm = 0; sm < 2; ++sm) {
      const __bf16* p = &sA[buf][wm * 32 + sm * 16 + l15][khalf * 8];
      v8bf lo = *reinterpret_cast<const v8bf*>(p);
      v8bf hi = *reinterpret_cast<const v8bf*>(p + 16);
#pragma unroll
      for (int i = 0; i < 8; ++i) { afr[sm][i] = lo[i]; afr[sm][8 + i] = hi[i]; }
    }
#pragma unroll
    for (int sn = 0; sn < 4; ++sn) {
      const __bf16* p = &sB[buf][wn * 64 + sn * 16 + l15][khalf * 8];
      v8bf lo = *reinterpret_cast<const v8bf*>(p);
      v8bf hi = *reinterpret_cast<const v8bf*>(p + 16);
#pragma unroll
      for (int i = 0; i < 8; ++i) { bfr[sn][i] = lo[i]; bfr[sn][8 + i] = hi[i]; }
    }

#pragma unroll
    for (int sm = 0; sm < 2; ++sm)
#pragma unroll
      for (int sn = 0; sn < 4; ++sn)
        acc[sm][sn] = __builtin_amdgcn_wmma_f32_16x16x32_bf16(
            false, afr[sm], false, bfr[sn], (short)0, acc[sm][sn], false, false);

    if (kt + 1 < NK) {
#if USE_ASYNC_LDS
      wait_asynccnt0();   // this wave's async tile (k+1) landed in LDS
      __syncthreads();    // all waves' tiles landed; prior reads of buf done
#else
      __syncthreads();
      store_lds(buf ^ 1);
      __syncthreads();
#endif
    }
  }

  // Epilogue: C/D layout is lane0-15 -> M = vgpr i, lane16-31 -> M = 8+i, N = lane&15
#pragma unroll
  for (int sm = 0; sm < 2; ++sm) {
#pragma unroll
    for (int sn = 0; sn < 4; ++sn) {
      const int ng = (int)n0 + wn * 64 + sn * 16 + l15;
      const int mgb = (int)m0 + wm * 32 + sm * 16 + khalf * 8;
      const float bi = bias[ng];
#pragma unroll
      for (int i = 0; i < 8; ++i) {
        float v = acc[sm][sn][i] + bi;
        v = v > 0.0f ? v : 0.0f;
        Z[(size_t)(mgb + i) * F + ng] = v;
      }
    }
  }
}

// ---------------------------------------------------------------------------
// Kernel 4: exact top-64 per row via 4-round radix select on float bits
// ---------------------------------------------------------------------------
__global__ __launch_bounds__(256) void sae_topk64(const float* __restrict__ Z,
                                                  float* __restrict__ vals,
                                                  int* __restrict__ idxs,
                                                  int F) {
  const int b = blockIdx.x;
  const float* zr = Z + (size_t)b * F;
  const int tid = threadIdx.x;

  __shared__ unsigned hist[256];
  __shared__ unsigned sPrefix;
  __shared__ int sKr;
  __shared__ int cG, cE;

  if (tid == 0) { sPrefix = 0u; sKr = TOPK; }
  __syncthreads();

  for (int shift = 24; shift >= 0; shift -= 8) {
    hist[tid] = 0u;
    __syncthreads();
    const unsigned prefix = sPrefix;
    for (int i = tid; i < F; i += 256) {
      unsigned u = __float_as_uint(zr[i]);
      bool cand = (shift == 24) || ((u >> (shift + 8)) == (prefix >> (shift + 8)));
      if (cand) atomicAdd(&hist[(u >> shift) & 255u], 1u);
    }
    __syncthreads();
    if (tid == 0) {
      unsigned above = 0;
      int sel = 0;
      const int Kr = sKr;
      for (int bin = 255; bin >= 0; --bin) {
        unsigned c = hist[bin];
        if (above + c >= (unsigned)Kr) { sel = bin; break; }
        above += c;
      }
      sKr = Kr - (int)above;                 // still needed inside selected bin
      sPrefix = prefix | ((unsigned)sel << shift);
    }
    __syncthreads();
  }

  const unsigned T = sPrefix;     // bits of the K-th largest value
  const int needEq = sKr;         // how many ties at T to take
  const int baseEq = TOPK - needEq;
  if (tid == 0) { cG = 0; cE = 0; }
  __syncthreads();

  for (int i = tid; i < F; i += 256) {
    float v = zr[i];
    unsigned u = __float_as_uint(v);
    if (u > T) {
      int s = atomicAdd(&cG, 1);
      vals[(size_t)b * TOPK + s] = v;
      idxs[(size_t)b * TOPK + s] = i;
    } else if (u == T) {
      int e = atomicAdd(&cE, 1);
      if (e < needEq) {
        int s = baseEq + e;
        vals[(size_t)b * TOPK + s] = v;
        idxs[(size_t)b * TOPK + s] = i;
      }
    }
  }
}

// ---------------------------------------------------------------------------
// Kernel 5: sparse decode  recon[b,:] = sum_k vals[b,k] * W_dec[idx[b,k],:] + b_dec
// W_dec (128 MB) is L2-resident on MI455X (192 MB L2); prefetch next gather row.
// ---------------------------------------------------------------------------
__global__ __launch_bounds__(256) void sae_decode(const float* __restrict__ vals,
                                                  const int* __restrict__ idxs,
                                                  const float* __restrict__ Wd,
                                                  const float* __restrict__ bd,
                                                  float* __restrict__ out,
                                                  int D) {
  const int b = blockIdx.x;
  const int tid = threadIdx.x;
  __shared__ float sv[TOPK];
  __shared__ int si[TOPK];
  if (tid < TOPK) {
    sv[tid] = vals[(size_t)b * TOPK + tid];
    si[tid] = idxs[(size_t)b * TOPK + tid];
  }
  __syncthreads();

  const int R = D >> 8;   // D / 256 columns per thread (D=2048 -> 8)
  float acc[8];
#pragma unroll
  for (int r = 0; r < 8; ++r) {
    if (r < R) acc[r] = bd[tid + (r << 8)];
  }
  for (int k = 0; k < TOPK; ++k) {
    const float v = sv[k];
    const float* wr = Wd + (size_t)si[k] * D;
    if (k + 1 < TOPK) {
      __builtin_prefetch(Wd + (size_t)si[k + 1] * D + tid, 0, 1);
    }
#pragma unroll
    for (int r = 0; r < 8; ++r) {
      if (r < R) acc[r] += v * wr[tid + (r << 8)];
    }
  }
#pragma unroll
  for (int r = 0; r < 8; ++r) {
    if (r < R) out[(size_t)b * D + tid + (r << 8)] = acc[r];
  }
}

// ---------------------------------------------------------------------------
// Host-side launch
// ---------------------------------------------------------------------------
extern "C" void kernel_launch(void* const* d_in, const int* in_sizes, int n_in,
                              void* d_out, int out_size, void* d_ws, size_t ws_size,
                              hipStream_t stream) {
  const float* x     = (const float*)d_in[0];
  const float* W_enc = (const float*)d_in[1];
  const float* b_enc = (const float*)d_in[2];
  const float* W_dec = (const float*)d_in[3];
  const float* b_dec = (const float*)d_in[4];

  const int D = in_sizes[4];                       // b_dec has D elements
  const int F = in_sizes[2];                       // b_enc has F elements
  const long long Bm = (long long)in_sizes[0] / D; // rows of x

  // workspace carve-out (256B aligned)
  char* wsb = (char*)d_ws;
  size_t off = 0;
  auto carve = [&](size_t bytes) -> char* {
    off = (off + 255) & ~(size_t)255;
    char* p = wsb + off;
    off += bytes;
    return p;
  };
  __bf16* xbf  = (__bf16*)carve((size_t)Bm * D * sizeof(__bf16));
  __bf16* wT   = (__bf16*)carve((size_t)F * D * sizeof(__bf16));
  float*  Z    = (float*)carve((size_t)Bm * F * sizeof(float));
  float*  vals = (float*)carve((size_t)Bm * TOPK * sizeof(float));
  int*    idxs = (int*)carve((size_t)Bm * TOPK * sizeof(int));
  (void)ws_size;

  // 1. x -> bf16
  {
    long long n4 = (Bm * (long long)D) / 4;
    int blocks = (int)((n4 + 255) / 256);
    sae_conv_x<<<blocks, 256, 0, stream>>>(x, xbf, n4);
  }
  // 2. W_enc -> W_encT bf16
  {
    dim3 g(F / 32, D / 32), blk(32, 8);
    sae_transpose_wenc<<<g, blk, 0, stream>>>(W_enc, wT, D, F);
  }
  // 3. encoder WMMA GEMM + bias + relu
  {
    dim3 g(F / TN, (unsigned)(Bm / TM));
    sae_enc_gemm<<<g, 256, 0, stream>>>(xbf, wT, b_enc, Z, D, F);
  }
  // 4. top-64 per row
  sae_topk64<<<(unsigned)Bm, 256, 0, stream>>>(Z, vals, idxs, F);
  // 5. sparse decode
  sae_decode<<<(unsigned)Bm, 256, 0, stream>>>(vals, idxs, W_dec, b_dec, (float*)d_out, D);
}
